// AdvancedLSTMForecaster_30270929502697
// MI455X (gfx1250) — compile-verified, matrix operand order checked by reading, not run on hardware
//
#include <hip/hip_runtime.h>
#include <hip/hip_bf16.h>

// ---------------------------------------------------------------------------
// Problem constants (from reference)
// ---------------------------------------------------------------------------
#define BB   128
#define TT   256
#define DIN  32
#define HH   256
#define LL   4
#define GG   1024   // 4*H
#define NHD  8
#define HD   32
#define NBLK 16
#define NTHR 512

typedef __attribute__((ext_vector_type(16))) __bf16 v16bf;
typedef __attribute__((ext_vector_type(8)))  float  v8f;
typedef __attribute__((ext_vector_type(4)))  int    v4i;

union FragU { v16bf v; unsigned u[8]; };
struct U16 { unsigned a, b, c, d; };

// ---------------------------------------------------------------------------
// CDNA5 feature probes
// ---------------------------------------------------------------------------
#if defined(__gfx1250__) && __has_builtin(__builtin_amdgcn_global_load_async_to_lds_b128)
#define HAS_ASYNC 1
#else
#define HAS_ASYNC 0
#endif

__device__ inline void copy16_g2l(void* ldst, const void* gsrc) {
#if HAS_ASYNC
  __builtin_amdgcn_global_load_async_to_lds_b128(
      (__attribute__((address_space(1))) v4i*)(unsigned long long)gsrc,
      (__attribute__((address_space(3))) v4i*)(unsigned)(unsigned long long)ldst,
      0, 0);
#else
  *(U16*)ldst = *(const U16*)gsrc;
#endif
}

__device__ inline void wait_async() {
#if HAS_ASYNC
# if __has_builtin(__builtin_amdgcn_s_wait_asynccnt)
  __builtin_amdgcn_s_wait_asynccnt(0);
# else
  asm volatile("s_wait_asynccnt 0" ::: "memory");
# endif
#endif
}

// ---------------------------------------------------------------------------
// bf16 helpers (storage = ushort; conversion manual RNE, no __bf16 arithmetic)
// ---------------------------------------------------------------------------
__device__ inline unsigned short f2bf(float f) {
  unsigned u = __float_as_uint(f);
  unsigned r = u + 0x7FFFu + ((u >> 16) & 1u);
  return (unsigned short)(r >> 16);
}
__device__ inline float bf2f(unsigned short h) {
  return __uint_as_float(((unsigned)h) << 16);
}
__device__ inline float sigm(float x) { return 1.f / (1.f + __expf(-x)); }

// ---------------------------------------------------------------------------
// WMMA fragment helpers (wave32; A 16x32 bf16 row-major, B pre-packed)
// A layout: lanes 0-15: M=lane, K pairs {0,2,4,6,16,18,20,22}; lanes 16-31: +8
// B is packed offline so lane L reads dwords [8L, 8L+8) of a 256-dword tile.
// ---------------------------------------------------------------------------
__device__ inline v16bf load_frag_a(const unsigned short* base, int ld) {
  const int lane = threadIdx.x & 31;
  const int m  = lane & 15;
  const int kb = (lane >> 4) << 3;
  const unsigned* row = (const unsigned*)(base + (size_t)m * ld);
  FragU f;
#pragma unroll
  for (int j = 0; j < 8; ++j) {
    int k = (j < 4) ? (kb + 2 * j) : (kb + 16 + 2 * (j - 4));
    f.u[j] = row[k >> 1];
  }
  return f.v;
}
__device__ inline v16bf load_frag_b(const unsigned* tile) {
  const int lane = threadIdx.x & 31;
  const unsigned* p = tile + lane * 8;
  FragU f;
#pragma unroll
  for (int j = 0; j < 8; ++j) f.u[j] = p[j];
  return f.v;
}
__device__ inline v8f wmma_bf16(v16bf a, v16bf b, v8f c) {
  return __builtin_amdgcn_wmma_f32_16x16x32_bf16(false, a, false, b, (short)0, c,
                                                 false, false);
}

// ---------------------------------------------------------------------------
// Workspace layout (bytes)
// ---------------------------------------------------------------------------
static constexpr size_t OFF_BAR   = 0;                                      // 2 u32 barrier state
static constexpr size_t OFF_SSLOT = 256;                                    // f32[T][L][B] batch-softmax scores
static constexpr size_t OFF_XT    = OFF_SSLOT + (size_t)TT * LL * BB * 4;   // bf16[T][B][DIN]
static constexpr size_t OFF_H     = OFF_XT + (size_t)TT * BB * DIN * 2;     // bf16[L][2][B][H] (t-parity dbl buf)
static constexpr size_t OFF_OUTS  = OFF_H + (size_t)LL * 2 * BB * HH * 2;   // bf16[T][B][H]
static constexpr size_t OFF_CWS   = OFF_OUTS + (size_t)TT * BB * HH * 2;    // f32[NBLK][L][B*16] c-state
static constexpr size_t OFF_WG    = OFF_CWS + (size_t)NBLK * LL * BB * 16 * 4;

__host__ __device__ constexpr size_t wgDwords(int l) { return (size_t)(l == 0 ? 9 : 16) * 64 * 256; }
__host__ __device__ constexpr size_t wgOff(int l) {
  size_t o = OFF_WG;
  for (int i = 0; i < l; ++i) o += wgDwords(i) * 4;
  return o;
}
static constexpr size_t OFF_HW = wgOff(LL);
__host__ __device__ constexpr size_t hwDwords(int l) { return (size_t)(l == 0 ? 1 : 8) * 16 * 256; }
__host__ __device__ constexpr size_t hwOff(int l, int which) {
  size_t o = OFF_HW;
  for (int i = 0; i < l; ++i) o += 2 * hwDwords(i) * 4;
  return o + (size_t)which * hwDwords(l) * 4;
}
static constexpr size_t MHA_DW    = (size_t)16 * 16 * 256;
static constexpr size_t OFF_WQ    = hwOff(LL, 0);
static constexpr size_t OFF_WK    = OFF_WQ + MHA_DW * 4;
static constexpr size_t OFF_WV    = OFF_WK + MHA_DW * 4;
static constexpr size_t OFF_WO    = OFF_WV + MHA_DW * 4;
static constexpr size_t OFF_KBUF  = OFF_WO + MHA_DW * 4;
static constexpr size_t OFF_VBUF  = OFF_KBUF + (size_t)TT * BB * HH * 2;
static constexpr size_t OFF_QLAST = OFF_VBUF + (size_t)TT * BB * HH * 2;
static constexpr size_t OFF_ATT   = OFF_QLAST + (size_t)BB * HH * 2;
static constexpr size_t OFF_FIN   = OFF_ATT + (size_t)BB * HH * 2;

// ---------------------------------------------------------------------------
// Grid-wide sense-reversing barrier (persistent 16-block LSTM kernel)
// ---------------------------------------------------------------------------
__device__ inline void grid_barrier(unsigned* cnt, unsigned* gen) {
  __syncthreads();
  if (threadIdx.x == 0) {
    unsigned g = __hip_atomic_load(gen, __ATOMIC_RELAXED, __HIP_MEMORY_SCOPE_AGENT);
    unsigned a = __hip_atomic_fetch_add(cnt, 1u, __ATOMIC_ACQ_REL, __HIP_MEMORY_SCOPE_AGENT);
    if (a == NBLK - 1) {
      __hip_atomic_store(cnt, 0u, __ATOMIC_RELAXED, __HIP_MEMORY_SCOPE_AGENT);
      __hip_atomic_fetch_add(gen, 1u, __ATOMIC_RELEASE, __HIP_MEMORY_SCOPE_AGENT);
    } else {
      while (__hip_atomic_load(gen, __ATOMIC_ACQUIRE, __HIP_MEMORY_SCOPE_AGENT) == g)
        __builtin_amdgcn_s_sleep(1);
    }
  }
  __syncthreads();
}

// ---------------------------------------------------------------------------
// Prep kernels
// ---------------------------------------------------------------------------
__global__ void k_init(char* __restrict__ ws) {
  const size_t n = OFF_WG / 4;  // zero barrier, score slots, xT, h, outs, c-state
  unsigned* p = (unsigned*)ws;
  for (size_t i = (size_t)blockIdx.x * blockDim.x + threadIdx.x; i < n;
       i += (size_t)gridDim.x * blockDim.x)
    p[i] = 0u;
}

__global__ void k_xt(const float* __restrict__ x, unsigned short* __restrict__ xt) {
  const size_t n = (size_t)BB * TT * DIN;  // x[B,T,D] -> xt[T,B,D] bf16
  for (size_t i = (size_t)blockIdx.x * blockDim.x + threadIdx.x; i < n;
       i += (size_t)gridDim.x * blockDim.x) {
    int d = (int)(i % DIN);
    size_t r = i / DIN;
    int t = (int)(r % TT);
    int b = (int)(r / TT);
    xt[((size_t)t * BB + b) * DIN + d] = f2bf(x[i]);
  }
}

// Pack row-major fp32 W[K,N] into per-tile WMMA-B dword layout (bf16 pairs).
__global__ void k_pack_b(const float* __restrict__ W, int K, int N,
                         unsigned* __restrict__ out) {
  const int total = (K * N) >> 1;
  const int ntiles = N >> 4;
  for (int idx = blockIdx.x * blockDim.x + threadIdx.x; idx < total;
       idx += gridDim.x * blockDim.x) {
    int tile = idx >> 8, rem = idx & 255;
    int lane = rem >> 3, j = rem & 7;
    int kt = tile / ntiles, nt = tile - kt * ntiles;
    int n  = (nt << 4) + (lane & 15);
    int kb = (lane >> 4) << 3;
    int kl = (j < 4) ? (kb + 2 * j) : (kb + 16 + 2 * (j - 4));
    int k  = (kt << 5) + kl;
    unsigned lo = f2bf(W[(size_t)k * N + n]);
    unsigned hi = f2bf(W[((size_t)k + 1) * N + n]);
    out[idx] = lo | (hi << 16);
  }
}

// ---------------------------------------------------------------------------
// Persistent recurrent kernel: 16 blocks x 512 threads, block b owns hidden
// columns [16b, 16b+16).  Grid-synced each layer-step.
// ---------------------------------------------------------------------------
struct LstmParams {
  const float* Wa[LL];
  const float* bg[LL];
  const float* bht[LL];
  const float* bhg[LL];
};

__global__ __launch_bounds__(NTHR, 1) void k_lstm(char* __restrict__ ws, LstmParams P) {
  __shared__ unsigned short Abuf[2][BB * 32];  // 16 KB double-buffered A chunk
  __shared__ float gbuf[4 * BB * 16];          // 32 KB: i,f,o,g planes (reused)
  __shared__ float attn[BB];
  __shared__ float red[2];

  const int tid  = threadIdx.x;
  const int blk  = blockIdx.x;
  const int wave = tid >> 5;
  const int lane = tid & 31;
  const int col  = lane & 15;
  const int rb   = (lane >> 4) << 3;

  unsigned* bar = (unsigned*)(ws + OFF_BAR);
  float* sslot  = (float*)(ws + OFF_SSLOT);
  const unsigned short* xt = (const unsigned short*)(ws + OFF_XT);
  unsigned short* hbase = (unsigned short*)(ws + OFF_H);
  unsigned short* outs  = (unsigned short*)(ws + OFF_OUTS);
  float* cws = (float*)(ws + OFF_CWS) + (size_t)blk * LL * BB * 16;

  for (int t = 0; t < TT; ++t) {
    const int pc = t & 1;        // parity of h(t)
    const int pp = (t + 1) & 1;  // parity of h(t-1)
    for (int l = 0; l < LL; ++l) {
      const int KT = (l == 0) ? 9 : 16;
      const unsigned* wg = (const unsigned*)(ws + wgOff(l));
      const unsigned short* hbelow =
          (l == 0) ? (xt + (size_t)t * BB * DIN)
                   : (hbase + ((size_t)((l - 1) * 2 + pc)) * BB * HH);
      const int belowCols = (l == 0) ? DIN : HH;
      const unsigned short* hprev = hbase + ((size_t)(l * 2 + pp)) * BB * HH;

      // ---- stage A chunk 0 (always the "below" input) ----
      {
        int row = tid >> 2, seg = tid & 3;
        copy16_g2l(&Abuf[0][row * 32 + seg * 8],
                   hbelow + (size_t)row * belowCols + seg * 8);
      }
      wait_async();
      __syncthreads();

      const int mt  = wave & 7;
      const int gp  = wave >> 3;          // gate pair: 0 -> {i,f}, 1 -> {o,g}
      const int nt0 = (gp * 2 + 0) * 16 + blk;
      const int nt1 = (gp * 2 + 1) * 16 + blk;
      v8f acc0 = {0.f, 0.f, 0.f, 0.f, 0.f, 0.f, 0.f, 0.f};
      v8f acc1 = acc0;

      // ---- gates GEMM: [128 x K] x [K x 1024], K chunked by 32 ----
      for (int kt = 0; kt < KT; ++kt) {
        const int cur = kt & 1;
        if (kt + 1 < KT) {
          int row = tid >> 2, seg = tid & 3;
          int kn = kt + 1;
          const unsigned short* src;
          int koff;
          if (l == 0)      { src = hprev;  koff = (kn - 1) * 32 + seg * 8; }
          else if (kn < 8) { src = hbelow; koff = kn * 32 + seg * 8; }
          else             { src = hprev;  koff = (kn - 8) * 32 + seg * 8; }
          copy16_g2l(&Abuf[1 - cur][row * 32 + seg * 8],
                     src + (size_t)row * HH + koff);
          __builtin_prefetch(wg + ((size_t)(kt + 1) * 64 + nt0) * 256, 0, 1);
        }
        v16bf a  = load_frag_a(&Abuf[cur][mt * 16 * 32], 32);
        v16bf b0 = load_frag_b(wg + ((size_t)kt * 64 + nt0) * 256);
        v16bf b1 = load_frag_b(wg + ((size_t)kt * 64 + nt1) * 256);
        acc0 = wmma_bf16(a, b0, acc0);
        acc1 = wmma_bf16(a, b1, acc1);
        if (kt + 1 < KT) wait_async();
        __syncthreads();
      }

      // ---- gate activations -> LDS planes ----
      {
        const float* bgp = P.bg[l];
        const float b0 = bgp[(gp * 2 + 0) * HH + blk * 16 + col];
        const float b1 = bgp[(gp * 2 + 1) * HH + blk * 16 + col];
#pragma unroll
        for (int v = 0; v < 8; ++v) {
          int row = mt * 16 + rb + v;
          float x0 = sigm(acc0[v] + b0);
          float x1 = acc1[v] + b1;
          x1 = (gp == 1) ? tanhf(x1) : sigm(x1);
          gbuf[(gp * 2 + 0) * BB * 16 + row * 16 + col] = x0;
          gbuf[(gp * 2 + 1) * BB * 16 + row * 16 + col] = x1;
        }
      }
      if (tid < BB) attn[tid] = 0.f;
      __syncthreads();

      // ---- c/h update + batch-attention score partials (waves 0..7) ----
      if (wave < 8) {
        const float wa = P.Wa[l][blk * 16 + col];
        float* cp = cws + (size_t)l * BB * 16;
#pragma unroll
        for (int v = 0; v < 8; ++v) {
          int row = mt * 16 + rb + v;
          int idx = row * 16 + col;
          float gi = gbuf[idx];
          float gf = gbuf[BB * 16 + idx];
          float go = gbuf[2 * BB * 16 + idx];
          float gg = gbuf[3 * BB * 16 + idx];
          float c  = gf * cp[idx] + gi * gg;
          cp[idx]  = c;
          float h  = go * tanhf(c);
          gbuf[3 * BB * 16 + idx] = h;  // stash h in dead g-plane
          atomicAdd(&attn[row], h * wa);
        }
      }
      __syncthreads();
      if (tid < BB)
        atomicAdd(&sslot[((size_t)t * LL + l) * BB + tid], attn[tid]);
      grid_barrier(bar, bar + 1);

      // ---- softmax over batch dim (ba cancels in softmax) ----
      if (tid < BB) attn[tid] = sslot[((size_t)t * LL + l) * BB + tid];
      __syncthreads();
      if (tid == 0) {
        float mx = -3.0e38f;
        for (int i = 0; i < BB; ++i) mx = fmaxf(mx, attn[i]);
        float sm = 0.f;
        for (int i = 0; i < BB; ++i) sm += __expf(attn[i] - mx);
        red[0] = mx;
        red[1] = sm;
      }
      __syncthreads();
      if (tid < BB) attn[tid] = __expf(attn[tid] - red[0]) / red[1];
      __syncthreads();

      // ---- highway transform/gate GEMMs (A straight from global) ----
      {
        const int which = wave >> 3;  // 0 = transform, 1 = gate
        const int KTH = (l == 0) ? 1 : 8;
        const unsigned* wb = (const unsigned*)(ws + hwOff(l, which));
        v8f acc = {0.f, 0.f, 0.f, 0.f, 0.f, 0.f, 0.f, 0.f};
        for (int kt = 0; kt < KTH; ++kt) {
          v16bf a = load_frag_a(hbelow + (size_t)(mt * 16) * belowCols + kt * 32,
                                belowCols);
          v16bf b = load_frag_b(wb + ((size_t)kt * 16 + blk) * 256);
          acc = wmma_bf16(a, b, acc);
        }
        const float* bias = which ? P.bhg[l] : P.bht[l];
        const float bs = bias[blk * 16 + col];
#pragma unroll
        for (int v = 0; v < 8; ++v) {
          int row = mt * 16 + rb + v;
          gbuf[which * BB * 16 + row * 16 + col] = acc[v] + bs;  // planes 0/1 dead
        }
      }
      __syncthreads();

      // ---- final combine + write h(t) bf16 (and outs for top layer) ----
      if (wave < 8) {
        unsigned short* hdst = hbase + ((size_t)(l * 2 + pc)) * BB * HH;
#pragma unroll
        for (int v = 0; v < 8; ++v) {
          int row = mt * 16 + rb + v;
          int idx = row * 16 + col;
          float htv = gbuf[idx];
          float hgv = sigm(gbuf[BB * 16 + idx]);
          float hv  = gbuf[3 * BB * 16 + idx] * attn[row];
          float hf  = hgv * htv + (1.f - hgv) * hv;
          unsigned short hb = f2bf(hf);
          hdst[(size_t)row * HH + blk * 16 + col] = hb;
          if (l == LL - 1)
            outs[((size_t)t * BB + row) * HH + blk * 16 + col] = hb;
        }
      }
      grid_barrier(bar, bar + 1);
    }
  }
}

// ---------------------------------------------------------------------------
// Generic M x 256 x 256 WMMA GEMM: out = bf16(A @ W + bias)
// ---------------------------------------------------------------------------
__global__ __launch_bounds__(256, 2) void k_proj(const unsigned short* __restrict__ A,
                                                 const unsigned* __restrict__ Bp,
                                                 const float* __restrict__ bias,
                                                 unsigned short* __restrict__ out) {
  __shared__ unsigned short Asm[16 * 256];  // 8 KB row strip
  const int tid = threadIdx.x;
  const size_t mbase = (size_t)blockIdx.x * 16;
  const unsigned short* Ab = A + mbase * 256;
  for (int i = tid; i < 512; i += 256)
    copy16_g2l(&((U16*)Asm)[i], &((const U16*)Ab)[i]);
  wait_async();
  __syncthreads();

  const int wave = tid >> 5, lane = tid & 31;
  const int col = lane & 15, rbv = (lane >> 4) << 3;
  const int nt0 = wave * 2, nt1 = wave * 2 + 1;
  v8f acc0 = {0.f, 0.f, 0.f, 0.f, 0.f, 0.f, 0.f, 0.f};
  v8f acc1 = acc0;
  for (int kt = 0; kt < 16; ++kt) {
    v16bf a  = load_frag_a(Asm + kt * 32, 256);
    v16bf b0 = load_frag_b(Bp + ((size_t)kt * 16 + nt0) * 256);
    v16bf b1 = load_frag_b(Bp + ((size_t)kt * 16 + nt1) * 256);
    acc0 = wmma_bf16(a, b0, acc0);
    acc1 = wmma_bf16(a, b1, acc1);
  }
  const float bs0 = bias[nt0 * 16 + col], bs1 = bias[nt1 * 16 + col];
#pragma unroll
  for (int v = 0; v < 8; ++v) {
    size_t row = mbase + rbv + v;
    out[row * 256 + nt0 * 16 + col] = f2bf(acc0[v] + bs0);
    out[row * 256 + nt1 * 16 + col] = f2bf(acc1[v] + bs1);
  }
}

// ---------------------------------------------------------------------------
// Single-query attention per (batch, head): only att[T-1] is needed.
// ---------------------------------------------------------------------------
__global__ __launch_bounds__(256) void k_attn(const unsigned short* __restrict__ q,
                                              const unsigned short* __restrict__ Kb,
                                              const unsigned short* __restrict__ Vb,
                                              unsigned short* __restrict__ out) {
  __shared__ float qv[HD];
  __shared__ float w[TT];
  __shared__ float red2[2];
  __shared__ float part[8][HD];
  const int b = blockIdx.x, h = blockIdx.y, tid = threadIdx.x;
  if (tid < HD) qv[tid] = bf2f(q[(size_t)b * HH + h * HD + tid]) * 0.17677669529f;
  __syncthreads();
  {
    const unsigned short* kr = Kb + ((size_t)tid * BB + b) * HH + h * HD;
    float s = 0.f;
#pragma unroll
    for (int d = 0; d < HD; ++d) s += qv[d] * bf2f(kr[d]);
    w[tid] = s;
  }
  __syncthreads();
  if (tid == 0) {
    float mx = -3.0e38f;
    for (int i = 0; i < TT; ++i) mx = fmaxf(mx, w[i]);
    float sm = 0.f;
    for (int i = 0; i < TT; ++i) sm += __expf(w[i] - mx);
    red2[0] = mx;
    red2[1] = sm;
  }
  __syncthreads();
  w[tid] = __expf(w[tid] - red2[0]) / red2[1];
  __syncthreads();
  {
    const int chunk = tid >> 5, d = tid & 31;
    float a = 0.f;
    for (int t = chunk * 32; t < chunk * 32 + 32; ++t)
      a += w[t] * bf2f(Vb[((size_t)t * BB + b) * HH + h * HD + d]);
    part[chunk][d] = a;
  }
  __syncthreads();
  if (tid < HD) {
    float r = 0.f;
#pragma unroll
    for (int c = 0; c < 8; ++c) r += part[c][tid];
    out[(size_t)b * HH + h * HD + tid] = f2bf(r);
  }
}

// ---------------------------------------------------------------------------
// Head MLP + eval-mode BatchNorm. One block per batch row.
// ---------------------------------------------------------------------------
__global__ __launch_bounds__(128) void k_head(const unsigned short* __restrict__ fin,
                                              const float* __restrict__ W1, const float* __restrict__ b1,
                                              const float* __restrict__ gamma, const float* __restrict__ beta,
                                              const float* __restrict__ W2, const float* __restrict__ b2,
                                              const float* __restrict__ W3, const float* __restrict__ b3,
                                              float* __restrict__ out) {
  __shared__ float y1[HH / 2];
  __shared__ float y2[HH / 4];
  const int row = blockIdx.x, tid = threadIdx.x;
  float a = 0.f;
  for (int k = 0; k < HH; ++k)
    a += bf2f(fin[(size_t)row * HH + k]) * W1[(size_t)k * (HH / 2) + tid];
  a = fmaxf(a + b1[tid], 0.f);
  a = a * 0.9999950000375f * gamma[tid] + beta[tid];  // 1/sqrt(1+1e-5)
  y1[tid] = a;
  __syncthreads();
  if (tid < HH / 4) {
    float s = 0.f;
    for (int k = 0; k < HH / 2; ++k) s += y1[k] * W2[(size_t)k * (HH / 4) + tid];
    y2[tid] = fmaxf(s + b2[tid], 0.f);
  }
  __syncthreads();
  if (tid == 0) {
    float s = 0.f;
    for (int k = 0; k < HH / 4; ++k) s += y2[k] * W3[k];
    out[row] = s + b3[0];
  }
}

// ---------------------------------------------------------------------------
// Launch: inputs follow setup_inputs() insertion order.
// ---------------------------------------------------------------------------
extern "C" void kernel_launch(void* const* d_in, const int* in_sizes, int n_in,
                              void* d_out, int out_size, void* d_ws, size_t ws_size,
                              hipStream_t stream) {
  char* ws = (char*)d_ws;
  const float* x = (const float*)d_in[0];
  LstmParams P;
  const float* Wg[LL];
  const float* Wht[LL];
  const float* Whg[LL];
  for (int l = 0; l < LL; ++l) {
    int b = 1 + 8 * l;  // Wg, bg, Wa, ba, Wht, bht, Whg, bhg
    Wg[l]    = (const float*)d_in[b + 0];
    P.bg[l]  = (const float*)d_in[b + 1];
    P.Wa[l]  = (const float*)d_in[b + 2];
    /* ba (d_in[b+3]) cancels inside softmax */
    Wht[l]   = (const float*)d_in[b + 4];
    P.bht[l] = (const float*)d_in[b + 5];
    Whg[l]   = (const float*)d_in[b + 6];
    P.bhg[l] = (const float*)d_in[b + 7];
  }
  const float* Wq = (const float*)d_in[33]; const float* bq = (const float*)d_in[34];
  const float* Wk = (const float*)d_in[35]; const float* bk = (const float*)d_in[36];
  const float* Wv = (const float*)d_in[37]; const float* bv = (const float*)d_in[38];
  const float* Wo = (const float*)d_in[39]; const float* bo = (const float*)d_in[40];
  const float* W1 = (const float*)d_in[41]; const float* b1 = (const float*)d_in[42];
  const float* gm = (const float*)d_in[43]; const float* bt = (const float*)d_in[44];
  const float* W2 = (const float*)d_in[45]; const float* b2 = (const float*)d_in[46];
  const float* W3 = (const float*)d_in[47]; const float* b3 = (const float*)d_in[48];

  k_init<<<512, 256, 0, stream>>>(ws);
  k_xt<<<1024, 256, 0, stream>>>(x, (unsigned short*)(ws + OFF_XT));

  auto pack = [&](const float* W, int K, int N, size_t off) {
    int dw = (K * N) / 2;
    int blocks = (dw + 255) / 256;
    if (blocks > 2048) blocks = 2048;
    k_pack_b<<<blocks, 256, 0, stream>>>(W, K, N, (unsigned*)(ws + off));
  };
  for (int l = 0; l < LL; ++l) {
    pack(Wg[l], (l == 0) ? (DIN + HH) : (2 * HH), GG, wgOff(l));
    pack(Wht[l], (l == 0) ? DIN : HH, HH, hwOff(l, 0));
    pack(Whg[l], (l == 0) ? DIN : HH, HH, hwOff(l, 1));
  }
  pack(Wq, HH, HH, OFF_WQ);
  pack(Wk, HH, HH, OFF_WK);
  pack(Wv, HH, HH, OFF_WV);
  pack(Wo, HH, HH, OFF_WO);

  k_lstm<<<NBLK, NTHR, 0, stream>>>(ws, P);

  unsigned short* outs = (unsigned short*)(ws + OFF_OUTS);
  k_proj<<<TT * BB / 16, 256, 0, stream>>>(outs, (unsigned*)(ws + OFF_WK), bk,
                                           (unsigned short*)(ws + OFF_KBUF));
  k_proj<<<TT * BB / 16, 256, 0, stream>>>(outs, (unsigned*)(ws + OFF_WV), bv,
                                           (unsigned short*)(ws + OFF_VBUF));
  k_proj<<<BB / 16, 256, 0, stream>>>(outs + (size_t)(TT - 1) * BB * HH,
                                      (unsigned*)(ws + OFF_WQ), bq,
                                      (unsigned short*)(ws + OFF_QLAST));
  k_attn<<<dim3(BB, NHD), 256, 0, stream>>>((const unsigned short*)(ws + OFF_QLAST),
                                            (const unsigned short*)(ws + OFF_KBUF),
                                            (const unsigned short*)(ws + OFF_VBUF),
                                            (unsigned short*)(ws + OFF_ATT));
  k_proj<<<BB / 16, 256, 0, stream>>>((const unsigned short*)(ws + OFF_ATT),
                                      (unsigned*)(ws + OFF_WO), bo,
                                      (unsigned short*)(ws + OFF_FIN));
  k_head<<<BB, 128, 0, stream>>>((const unsigned short*)(ws + OFF_FIN),
                                 W1, b1, gm, bt, W2, b2, W3, b3, (float*)d_out);
}
